// C2D_34419867910289
// MI455X (gfx1250) — matrix operand order repaired and checked by reading, not compiled, not run on hardware
//
#include <hip/hip_runtime.h>
#include <hip/hip_bf16.h>

#define BB   4096
#define NCC  32
#define DD   128
#define CCC  256
#define HHH  256

typedef __attribute__((ext_vector_type(16))) __bf16        bf16x16;
typedef __attribute__((ext_vector_type(8)))  float         floatx8;
typedef __attribute__((ext_vector_type(4)))  unsigned int  uint32x4;

union Frag {
  bf16x16      v;
  uint32x4     q[2];
};

__device__ inline __bf16 cvtbf(float x) { return (__bf16)x; }

// A-fragment (16x32 bf16): lane L: row m = L%16; element e -> k = (e>=8)*16 + (L/16)*8 + (e&7)
__device__ inline bf16x16 loadA_f32(const float* __restrict__ rowp, int k0) {
  bf16x16 v;
#pragma unroll
  for (int i = 0; i < 8; i++) v[i]     = cvtbf(rowp[k0 + i]);
#pragma unroll
  for (int i = 0; i < 8; i++) v[8 + i] = cvtbf(rowp[k0 + 16 + i]);
  return v;
}

__device__ inline bf16x16 loadA_lds(const __bf16* scr, int row, int k0) {
  Frag f;
  f.q[0] = *((const uint32x4*)(scr + row * 256 + k0));
  f.q[1] = *((const uint32x4*)(scr + row * 256 + k0 + 16));
  return f.v;
}

// B-fragment (32x16 bf16): lane L: col n = L%16; element e -> k = (e>=8)*16 + (L/16)*8 + (e&7)
// base is row-major [rows][rowlen] bf16 with "row" = N index, elements = K contiguous.
__device__ inline bf16x16 loadB(const __bf16* __restrict__ base, int row, int rowlen, int k0) {
  Frag f;
  const uint32x4* p = (const uint32x4*)(base + (size_t)row * rowlen + k0);
  f.q[0] = p[0];
  f.q[1] = p[2];
  return f.v;
}

__device__ inline floatx8 wmma_bf16(bf16x16 a, bf16x16 b, floatx8 c) {
  return __builtin_amdgcn_wmma_f32_16x16x32_bf16(false, a, false, b, (short)0, c, false, false);
}

__device__ inline float redsum16(float v) {
#pragma unroll
  for (int m = 8; m >= 1; m >>= 1) v += __shfl_xor(v, m, 16);
  return v;
}
__device__ inline float redmax16(float v) {
#pragma unroll
  for (int m = 8; m >= 1; m >>= 1) v = fmaxf(v, __shfl_xor(v, m, 16));
  return v;
}

// ---------------- Prep: K = E*Wk (bf16 [j][c][d]), V^T = (E*Wv)^T (bf16 [j][d][c]) --------
__global__ __launch_bounds__(128) void kv_prep(const float* __restrict__ emb,
                                               const float* __restrict__ Wk,
                                               const float* __restrict__ Wv,
                                               __bf16* __restrict__ Kbf,
                                               __bf16* __restrict__ Vt) {
  int blk = blockIdx.x;
  int j = blk >> 8;         // / CCC
  int c = blk & 255;        // % CCC
  int d = threadIdx.x;      // 0..127
  __shared__ float e[DD];
  e[d] = emb[((size_t)j * CCC + c) * DD + d];
  __syncthreads();
  const float* wkp = Wk + (size_t)j * DD * DD + d;
  const float* wvp = Wv + (size_t)j * DD * DD + d;
  float kk = 0.f, vv = 0.f;
#pragma unroll 4
  for (int t = 0; t < DD; t++) {
    float et = e[t];
    kk += et * wkp[t * DD];
    vv += et * wvp[t * DD];
  }
  Kbf[((size_t)j * CCC + c) * DD + d] = cvtbf(kk);
  Vt [((size_t)j * DD + d) * CCC + c] = cvtbf(vv);
}

// ------- Prep: transpose + scale + f32->bf16  src [j][R][Cc] -> dst [j][Cc][R] -----------
__global__ __launch_bounds__(256) void tr_cvt(const float* __restrict__ src,
                                              __bf16* __restrict__ dst,
                                              int R, int Cc, float scale) {
  int j = blockIdx.x;
  int total = R * Cc;
  const float* s = src + (size_t)j * total;
  __bf16* d = dst + (size_t)j * total;
  for (int idx = threadIdx.x; idx < total; idx += blockDim.x) {
    int r = idx / Cc, c = idx - r * Cc;
    d[c * R + r] = cvtbf(s[idx] * scale);
  }
}

// ---------------- Fused main: per wave = 16 batch rows x 1 category ----------------------
__global__ __launch_bounds__(256) void fused_main(
    const float* __restrict__ cat,        // [B][NC][D] f32
    const __bf16* __restrict__ Kbf,       // [NC][C][D] bf16
    const __bf16* __restrict__ Vt,        // [NC][D][C] bf16
    const __bf16* __restrict__ Wqt,       // [NC][D(out)][D(in)] bf16 (pre-scaled 1/sqrt(D))
    const __bf16* __restrict__ W1t,       // [NC][H][D] bf16
    const __bf16* __restrict__ W2t,       // [NC][D][H] bf16
    const float* __restrict__ ln1g, const float* __restrict__ ln1b,
    const float* __restrict__ b1,
    const float* __restrict__ b2,
    const float* __restrict__ ln2g, const float* __restrict__ ln2b,
    const float* __restrict__ Wsw, const float* __restrict__ bsb,
    float* __restrict__ out) {
  __shared__ __bf16 scr_all[8 * 16 * 256];  // 64KB, wave-private slices

  const int j    = blockIdx.y;
  const int wave = threadIdx.x >> 5;
  const int lane = threadIdx.x & 31;
  const int g    = lane >> 4;       // half-wave group
  const int n    = lane & 15;       // A row / B col / C col index
  const int m0   = blockIdx.x * 128 + wave * 16;
  __bf16* scr = scr_all + wave * (16 * 256);

  const __bf16* wqb = Wqt + (size_t)j * DD * DD;
  const __bf16* kb  = Kbf + (size_t)j * CCC * DD;
  const __bf16* vb  = Vt  + (size_t)j * DD * CCC;
  const __bf16* w1b = W1t + (size_t)j * HHH * DD;
  const __bf16* w2b = W2t + (size_t)j * DD * HHH;
  __builtin_prefetch(kb, 0, 1);
  __builtin_prefetch(vb, 0, 1);
  __builtin_prefetch(w1b, 0, 1);

  const floatx8 fzero = {0.f, 0.f, 0.f, 0.f, 0.f, 0.f, 0.f, 0.f};

  // ---- Stage 1: q = X @ (Wq/sqrt(D))  (A direct from global f32, B from bf16 weights) ----
  const float* xrow = cat + ((size_t)(m0 + n) * NCC + j) * DD;
  floatx8 qAcc[8];
#pragma unroll
  for (int t = 0; t < 8; t++) qAcc[t] = fzero;
#pragma unroll
  for (int kk = 0; kk < 4; kk++) {
    bf16x16 a = loadA_f32(xrow, kk * 32 + g * 8);
#pragma unroll
    for (int t = 0; t < 8; t++) {
      bf16x16 b = loadB(wqb, t * 16 + n, DD, kk * 32 + g * 8);
      qAcc[t] = wmma_bf16(a, b, qAcc[t]);
    }
  }
#pragma unroll
  for (int t = 0; t < 8; t++)
#pragma unroll
    for (int r = 0; r < 8; r++)
      scr[(g * 8 + r) * 256 + t * 16 + n] = cvtbf(qAcc[t][r]);

  // ---- Stage 2: scores = q @ K^T ; softmax over C=256 ----
  floatx8 sAcc[16];
#pragma unroll
  for (int t = 0; t < 16; t++) sAcc[t] = fzero;
#pragma unroll
  for (int kk = 0; kk < 4; kk++) {
    bf16x16 a = loadA_lds(scr, n, kk * 32 + g * 8);
#pragma unroll
    for (int t = 0; t < 16; t++) {
      bf16x16 b = loadB(kb, t * 16 + n, DD, kk * 32 + g * 8);
      sAcc[t] = wmma_bf16(a, b, sAcc[t]);
    }
  }
#pragma unroll
  for (int r = 0; r < 8; r++) {
    float mx = sAcc[0][r];
#pragma unroll
    for (int t = 1; t < 16; t++) mx = fmaxf(mx, sAcc[t][r]);
    mx = redmax16(mx);
    float sum = 0.f;
#pragma unroll
    for (int t = 0; t < 16; t++) {
      float e = __expf(sAcc[t][r] - mx);
      sAcc[t][r] = e;
      sum += e;
    }
    sum = redsum16(sum);
    float inv = 1.0f / sum;
#pragma unroll
    for (int t = 0; t < 16; t++)
      scr[(g * 8 + r) * 256 + t * 16 + n] = cvtbf(sAcc[t][r] * inv);
  }

  // ---- Stage 3: h = alpha @ V ; residual + LN1 ----
  floatx8 hAcc[8];
#pragma unroll
  for (int t = 0; t < 8; t++) hAcc[t] = fzero;
#pragma unroll
  for (int kk = 0; kk < 8; kk++) {
    bf16x16 a = loadA_lds(scr, n, kk * 32 + g * 8);
#pragma unroll
    for (int t = 0; t < 8; t++) {
      bf16x16 b = loadB(vb, t * 16 + n, CCC, kk * 32 + g * 8);
      hAcc[t] = wmma_bf16(a, b, hAcc[t]);
    }
  }
  float g1[8], bb1[8];
#pragma unroll
  for (int t = 0; t < 8; t++) { g1[t] = ln1g[t * 16 + n]; bb1[t] = ln1b[t * 16 + n]; }
#pragma unroll
  for (int r = 0; r < 8; r++) {
    int brow = m0 + g * 8 + r;
    const float* xr = cat + ((size_t)brow * NCC + j) * DD;
    float s[8];
    float mean = 0.f;
#pragma unroll
    for (int t = 0; t < 8; t++) { s[t] = xr[t * 16 + n] + hAcc[t][r]; mean += s[t]; }
    mean = redsum16(mean) * (1.0f / 128.0f);
    float var = 0.f;
#pragma unroll
    for (int t = 0; t < 8; t++) { float d0 = s[t] - mean; var += d0 * d0; }
    var = redsum16(var) * (1.0f / 128.0f);
    float rs = rsqrtf(var + 1e-5f);
#pragma unroll
    for (int t = 0; t < 8; t++) {
      float y = (s[t] - mean) * rs * g1[t] + bb1[t];
      hAcc[t][r] = y;  // keep f32 for second residual
      scr[(g * 8 + r) * 256 + t * 16 + n] = cvtbf(y);
    }
  }

  // ---- Stage 4: ff1 = relu(h1 @ W1 + b1) ----
  floatx8 fAcc[16];
#pragma unroll
  for (int t = 0; t < 16; t++) fAcc[t] = fzero;
#pragma unroll
  for (int kk = 0; kk < 4; kk++) {
    bf16x16 a = loadA_lds(scr, n, kk * 32 + g * 8);
#pragma unroll
    for (int t = 0; t < 16; t++) {
      bf16x16 b = loadB(w1b, t * 16 + n, DD, kk * 32 + g * 8);
      fAcc[t] = wmma_bf16(a, b, fAcc[t]);
    }
  }
#pragma unroll
  for (int t = 0; t < 16; t++) {
    float bias = b1[(size_t)j * HHH + t * 16 + n];
#pragma unroll
    for (int r = 0; r < 8; r++) {
      float v = fAcc[t][r] + bias;
      v = v > 0.f ? v : 0.f;
      scr[(g * 8 + r) * 256 + t * 16 + n] = cvtbf(v);
    }
  }

  // ---- Stage 5: ff2 = relu @ W2 + b2 ; residual + LN2 ; sigmoid(h.Ws + bs) ----
  floatx8 oAcc[8];
#pragma unroll
  for (int t = 0; t < 8; t++) oAcc[t] = fzero;
#pragma unroll
  for (int kk = 0; kk < 8; kk++) {
    bf16x16 a = loadA_lds(scr, n, kk * 32 + g * 8);
#pragma unroll
    for (int t = 0; t < 8; t++) {
      bf16x16 b = loadB(w2b, t * 16 + n, HHH, kk * 32 + g * 8);
      oAcc[t] = wmma_bf16(a, b, oAcc[t]);
    }
  }
  float g2[8], bg2[8], b2v[8], wsv[8];
#pragma unroll
  for (int t = 0; t < 8; t++) {
    g2[t]  = ln2g[t * 16 + n];
    bg2[t] = ln2b[t * 16 + n];
    b2v[t] = b2 [(size_t)j * DD + t * 16 + n];
    wsv[t] = Wsw[(size_t)j * DD + t * 16 + n];
  }
  float bsj = bsb[j];
#pragma unroll
  for (int r = 0; r < 8; r++) {
    float s[8];
    float mean = 0.f;
#pragma unroll
    for (int t = 0; t < 8; t++) { s[t] = hAcc[t][r] + oAcc[t][r] + b2v[t]; mean += s[t]; }
    mean = redsum16(mean) * (1.0f / 128.0f);
    float var = 0.f;
#pragma unroll
    for (int t = 0; t < 8; t++) { float d0 = s[t] - mean; var += d0 * d0; }
    var = redsum16(var) * (1.0f / 128.0f);
    float rs = rsqrtf(var + 1e-5f);
    float dot = 0.f;
#pragma unroll
    for (int t = 0; t < 8; t++) {
      float y = (s[t] - mean) * rs * g2[t] + bg2[t];
      dot += y * wsv[t];
    }
    dot = redsum16(dot);
    if (n == 0) {
      float z = dot + bsj;
      out[(size_t)(m0 + g * 8 + r) * NCC + j] = 1.0f / (1.0f + __expf(-z));
    }
  }
}

extern "C" void kernel_launch(void* const* d_in, const int* in_sizes, int n_in,
                              void* d_out, int out_size, void* d_ws, size_t ws_size,
                              hipStream_t stream) {
  (void)in_sizes; (void)n_in; (void)out_size; (void)ws_size;
  const float* cat  = (const float*)d_in[1];
  const float* emb  = (const float*)d_in[2];
  const float* Wq   = (const float*)d_in[3];
  const float* Wk   = (const float*)d_in[4];
  const float* Wv   = (const float*)d_in[5];
  const float* ln1g = (const float*)d_in[6];
  const float* ln1b = (const float*)d_in[7];
  const float* W1   = (const float*)d_in[8];
  const float* b1   = (const float*)d_in[9];
  const float* W2   = (const float*)d_in[10];
  const float* b2   = (const float*)d_in[11];
  const float* ln2g = (const float*)d_in[12];
  const float* ln2b = (const float*)d_in[13];
  const float* Ws   = (const float*)d_in[14];
  const float* bs   = (const float*)d_in[15];
  float* out = (float*)d_out;

  // Workspace carve (bf16 elements)
  __bf16* Kbf = (__bf16*)d_ws;
  __bf16* Vt  = Kbf + (size_t)NCC * CCC * DD;
  __bf16* Wqt = Vt  + (size_t)NCC * DD * CCC;
  __bf16* W1t = Wqt + (size_t)NCC * DD * DD;
  __bf16* W2t = W1t + (size_t)NCC * HHH * DD;

  const float qscale = 0.08838834764831845f;  // 1/sqrt(128)
  kv_prep<<<dim3(NCC * CCC), 128, 0, stream>>>(emb, Wk, Wv, Kbf, Vt);
  tr_cvt<<<dim3(NCC), 256, 0, stream>>>(Wq, Wqt, DD, DD, qscale);
  tr_cvt<<<dim3(NCC), 256, 0, stream>>>(W1, W1t, DD, HHH, 1.0f);
  tr_cvt<<<dim3(NCC), 256, 0, stream>>>(W2, W2t, HHH, DD, 1.0f);

  fused_main<<<dim3(BB / 128, NCC), 256, 0, stream>>>(
      cat, Kbf, Vt, Wqt, W1t, W2t,
      ln1g, ln1b, b1, b2, ln2g, ln2b, Ws, bs, out);
}
